// CrossAttention_4294967296409
// MI455X (gfx1250) — matrix operand order repaired and checked by reading, not compiled, not run on hardware
//
#include <hip/hip_runtime.h>
#include <hip/hip_bf16.h>

typedef _Float16 f16;
typedef _Float16 h4  __attribute__((ext_vector_type(4)));
typedef _Float16 h8  __attribute__((ext_vector_type(8)));
typedef _Float16 v16h __attribute__((ext_vector_type(16)));
typedef float    v8f  __attribute__((ext_vector_type(8)));
typedef unsigned int v4u __attribute__((ext_vector_type(4)));
typedef int      v8i  __attribute__((ext_vector_type(8)));

// ---------------------------------------------------------------------------
// helpers
// ---------------------------------------------------------------------------
__device__ __forceinline__ v16h ld2(const f16* p0, const f16* p1) {
    union { v16h v; h8 h[2]; } u;
    u.h[0] = *(const h8*)p0;
    u.h[1] = *(const h8*)p1;
    return u.v;
}

__device__ __forceinline__ float rmax16(float v) {
    #pragma unroll
    for (int m = 1; m < 16; m <<= 1) v = fmaxf(v, __shfl_xor(v, m, 32));
    return v;
}
__device__ __forceinline__ float rsum16(float v) {
    #pragma unroll
    for (int m = 1; m < 16; m <<= 1) v += __shfl_xor(v, m, 32);
    return v;
}
__device__ __forceinline__ float rsum32(float v) {
    #pragma unroll
    for (int m = 1; m < 32; m <<= 1) v += __shfl_xor(v, m, 32);
    return v;
}

// LDS byte offset of a generic pointer derived from __shared__:
// on AMDGPU the generic (flat) address of LDS is {aperture_hi32, offset}.
__device__ __forceinline__ unsigned lds_off(const void* p) {
    return (unsigned)(unsigned long long)(uintptr_t)p;
}

// ---------------------------------------------------------------------------
// Tensor Data Mover: 2D tile load Global -> LDS (CDNA5 TDM, ISA ch. 8).
// data_size = 8 bytes.  tw = tile width in 8B elements (<= 65535),
// th = tile height (rows), stride = row stride in 8B elements.
// Rows are packed contiguously into LDS starting at ldsoff.
// Issued per-wave (EXEC ignored); caller must s_wait_tensorcnt.
// ---------------------------------------------------------------------------
__device__ __forceinline__ void tdm_load_2d(const void* gptr, unsigned ldsoff,
                                            unsigned tw, unsigned th,
                                            unsigned long long stride)
{
    const unsigned long long ga = (unsigned long long)(uintptr_t)gptr;
    v4u g0;
    g0[0] = 1u;                                      // count=1, user descriptor
    g0[1] = ldsoff;                                  // lds_addr (bytes)
    g0[2] = (unsigned)ga;                            // global_addr[31:0]
    g0[3] = ((unsigned)(ga >> 32) & 0x1FFFFFFu)      // global_addr[56:32]
          | 0x80000000u;                             // type = 2 ("image")
    v8i g1;
    g1[0] = (int)0x00030000u;                        // wg_mask=0, data_size=8B
    g1[1] = (int)((tw & 0xFFFFu) << 16);             // tensor_dim0[15:0]
    g1[2] = (int)((tw >> 16) | ((th & 0xFFFFu) << 16));   // dim0[31:16] | dim1[15:0]
    g1[3] = (int)((th >> 16) | ((tw & 0xFFFFu) << 16));   // dim1[31:16] | tile_dim0
    g1[4] = (int)(th & 0xFFFFu);                     // tile_dim1 (tile_dim2 = 0)
    g1[5] = (int)(unsigned)(stride & 0xFFFFFFFFu);   // tensor_dim0_stride[31:0]
    g1[6] = (int)(unsigned)((stride >> 32) & 0xFFFFu);    // stride[47:32]
    g1[7] = 0;                                       // tensor_dim1_stride = 0
    asm volatile("tensor_load_to_lds %0, %1" :: "s"(g0), "s"(g1) : "memory");
}

// ---------------------------------------------------------------------------
// LayerNorm (f32 in) -> f16 out.  One wave32 per row.
// ---------------------------------------------------------------------------
__global__ __launch_bounds__(256) void ln_f16_kernel(
    const float* __restrict__ x, const float* __restrict__ w,
    const float* __restrict__ b, f16* __restrict__ o, int rows, int dim)
{
    const int wv = threadIdx.x >> 5, lane = threadIdx.x & 31;
    const int row = blockIdx.x * 8 + wv;
    if (row >= rows) return;
    const float4* xr = (const float4*)(x + (size_t)row * dim);
    const int nv = dim >> 2;
    float s = 0.f, s2 = 0.f;
    for (int i = lane; i < nv; i += 32) {
        float4 v = xr[i];
        s  += v.x + v.y + v.z + v.w;
        s2 += v.x*v.x + v.y*v.y + v.z*v.z + v.w*v.w;
    }
    s  = rsum32(s);
    s2 = rsum32(s2);
    const float mean = s / dim;
    const float var  = s2 / dim - mean * mean;
    const float rstd = rsqrtf(var + 1e-5f);
    h4* orow = (h4*)(o + (size_t)row * dim);
    const float4* wv4 = (const float4*)w;
    const float4* bv4 = (const float4*)b;
    for (int i = lane; i < nv; i += 32) {
        float4 v = xr[i]; float4 ww = wv4[i]; float4 bb = bv4[i];
        h4 hv;
        hv[0] = (f16)((v.x - mean) * rstd * ww.x + bb.x);
        hv[1] = (f16)((v.y - mean) * rstd * ww.y + bb.y);
        hv[2] = (f16)((v.z - mean) * rstd * ww.z + bb.z);
        hv[3] = (f16)((v.w - mean) * rstd * ww.w + bb.w);
        orow[i] = hv;
    }
}

// ---------------------------------------------------------------------------
// f32 -> f16 converter (vectorized by 4)
// ---------------------------------------------------------------------------
__global__ __launch_bounds__(256) void cvt_f16_kernel(
    const float* __restrict__ s, f16* __restrict__ d, int n4)
{
    int i = blockIdx.x * blockDim.x + threadIdx.x;
    if (i >= n4) return;
    float4 v = ((const float4*)s)[i];
    h4 hv;
    hv[0] = (f16)v.x; hv[1] = (f16)v.y; hv[2] = (f16)v.z; hv[3] = (f16)v.w;
    ((h4*)d)[i] = hv;
}

// ---------------------------------------------------------------------------
// Tiled WMMA GEMM: C[M,N] = A[M,K] (f16, row-major) x B[K,N] (f16, row-major)
// Block: 256 thr (8 waves).  Tile 128x128, BK=32.  Wave tile 64x32 -> 8 WMMA.
// A tile staged by the Tensor Data Mover (wave 0 issues a 2D descriptor),
// B tile transposed into LDS [N][K] by all threads.
// MODE 0: f16 C row-major           (out0)
// MODE 1: q epilogue  (scale 1/8, reshape [16384,512] -> [b,h,s,d])
// MODE 2: kv epilogue (split: K -> [b,h,mn,d] in out0, V -> [b,h,d,mn] in out1)
// MODE 3: f32 C row-major           (out0)
// ---------------------------------------------------------------------------
template<int MODE>
__global__ __launch_bounds__(256) void gemm_wmma(
    const f16* __restrict__ A, const f16* __restrict__ B,
    void* __restrict__ out0, void* __restrict__ out1,
    int M, int N, int K)
{
    __shared__ __attribute__((aligned(16))) f16 As[128 * 32];
    __shared__ __attribute__((aligned(16))) f16 Bs[128 * 32];   // stored [N][K]

    const int tid  = threadIdx.x;
    const int lane = tid & 31, w = tid >> 5;
    const int g = lane >> 4, lr = lane & 15;
    const int wm = w >> 2, wn = w & 3;          // 2 x 4 wave grid
    const int m0 = blockIdx.y * 128, n0 = blockIdx.x * 128;

    v8f acc[4][2];
    #pragma unroll
    for (int mi = 0; mi < 4; ++mi)
        #pragma unroll
        for (int ni = 0; ni < 2; ++ni) { v8f z = {}; acc[mi][ni] = z; }

    for (int kt = 0; kt < K; kt += 32) {
        // ---- stage A tile [128][32] via TDM: 128 rows x 8 (8B) elements,
        //      row stride K/4 (8B) elements
        if (w == 0) {
            tdm_load_2d(A + (size_t)m0 * K + kt, lds_off(As),
                        8u, 128u, (unsigned long long)(K >> 2));
        }
        // ---- stage B tile transposed -> Bs[N][K] (all 256 threads)
        {
            const int kk = tid >> 3, cc = tid & 7;
            const f16* src = B + (size_t)(kt + kk) * N + n0 + cc * 16;
            h8 b0 = *(const h8*)src;
            h8 b1 = *(const h8*)(src + 8);
            #pragma unroll
            for (int j = 0; j < 8; ++j) {
                Bs[(cc * 16 + j) * 32 + kk]     = b0[j];
                Bs[(cc * 16 + 8 + j) * 32 + kk] = b1[j];
            }
        }
        if (kt + 32 < K) {
            __builtin_prefetch(B + (size_t)(kt + 32 + (tid >> 3)) * N + n0 + (tid & 7) * 16, 0, 1);
        }
        if (w == 0) __builtin_amdgcn_s_wait_tensorcnt(0);
        __syncthreads();

        // ---- compute
        v16h bf[2];
        #pragma unroll
        for (int ni = 0; ni < 2; ++ni) {
            const f16* p = &Bs[(wn * 32 + ni * 16 + lr) * 32 + 8 * g];
            bf[ni] = ld2(p, p + 16);
        }
        #pragma unroll
        for (int mi = 0; mi < 4; ++mi) {
            const f16* p = &As[(wm * 64 + mi * 16 + lr) * 32 + 8 * g];
            v16h af = ld2(p, p + 16);
            #pragma unroll
            for (int ni = 0; ni < 2; ++ni)
                acc[mi][ni] = __builtin_amdgcn_wmma_f32_16x16x32_f16(
                    false, af, false, bf[ni], (short)0, acc[mi][ni], false, false);
        }
        __syncthreads();
    }

    // ---- epilogue
    #pragma unroll
    for (int mi = 0; mi < 4; ++mi) {
        #pragma unroll
        for (int ni = 0; ni < 2; ++ni) {
            #pragma unroll
            for (int r = 0; r < 8; ++r) {
                const int gm = m0 + wm * 64 + mi * 16 + r + 8 * g;
                const int gn = n0 + wn * 32 + ni * 16 + lr;
                const float c = acc[mi][ni][r];
                if constexpr (MODE == 0) {
                    ((f16*)out0)[(size_t)gm * N + gn] = (f16)c;
                } else if constexpr (MODE == 1) {
                    // M = b*4096 + i ; N = h*64 + d ; q scaled by 1/sqrt(64)
                    const int bb = gm >> 12, i = gm & 4095;
                    const int h = gn >> 6,  d = gn & 63;
                    ((f16*)out0)[(((size_t)(bb * 8 + h)) * 4096 + i) * 64 + d] =
                        (f16)(c * 0.125f);
                } else if constexpr (MODE == 2) {
                    // M = b*512 + j ; N<512 -> K[b,h,j,d] ; N>=512 -> V^T[b,h,d,j]
                    const int bb = gm >> 9, j = gm & 511;
                    if (gn < 512) {
                        const int h = gn >> 6, d = gn & 63;
                        ((f16*)out0)[(((size_t)(bb * 8 + h)) * 512 + j) * 64 + d] = (f16)c;
                    } else {
                        const int c2 = gn - 512, h = c2 >> 6, d = c2 & 63;
                        ((f16*)out1)[(((size_t)(bb * 8 + h)) * 64 + d) * 512 + j] = (f16)c;
                    }
                } else {
                    ((float*)out0)[(size_t)gm * N + gn] = c;
                }
            }
        }
    }
}

// ---------------------------------------------------------------------------
// Flash attention with image-causal block mask.
// Grid: (b*h = 32, s/128 = 32), block 256 (8 waves, 16 queries each).
// LDS: K[512][64] + V^T[64][512] (both staged by TDM) + per-wave P scratch.
// Each 32-key block belongs to exactly one image (img = jb/2), so the mask
// is a per-row compare against media_locations.
// ---------------------------------------------------------------------------
__global__ __launch_bounds__(256) void attn_kernel(
    const f16* __restrict__ qh,    // [b,h,4096,64] (pre-scaled by 1/8)
    const f16* __restrict__ kh,    // [b,h,512,64]
    const f16* __restrict__ vth,   // [b,h,64,512]
    const int* __restrict__ locs,  // [b,4096]
    f16* __restrict__ aout)        // [b,4096,512]  (inner = h*64+d)
{
    extern __shared__ __align__(16) f16 smem[];
    f16* ktl = smem;               // 512*64
    f16* vtl = smem + 512 * 64;    // 64*512
    f16* pba = smem + 2 * 512 * 64;// 8 * 512

    const int tid = threadIdx.x, lane = tid & 31, w = tid >> 5;
    const int g = lane >> 4, lr = lane & 15;
    const int bh = blockIdx.x, bb = bh >> 3, h = bh & 7;
    const int i0 = blockIdx.y * 128 + w * 16;

    // stage K and V^T for this (b,h): two 64 KB TDM transfers issued by wave 0
    if (w == 0) {
        tdm_load_2d(kh  + (size_t)bh * 512 * 64, lds_off(ktl), 8192u, 1u, 8192ull);
        tdm_load_2d(vth + (size_t)bh * 64 * 512, lds_off(vtl), 8192u, 1u, 8192ull);
        __builtin_amdgcn_s_wait_tensorcnt(0);
    }
    __syncthreads();

    // q fragments for this wave's 16 queries (A layout, two K-halves of d=64)
    const f16* qb = qh + ((size_t)bh * 4096 + i0 + lr) * 64;
    const v16h aq0 = ld2(qb + 8 * g,      qb + 8 * g + 16);
    const v16h aq1 = ld2(qb + 32 + 8 * g, qb + 32 + 8 * g + 16);

    int locv[8];
    #pragma unroll
    for (int r = 0; r < 8; ++r)
        locv[r] = locs[bb * 4096 + i0 + r + 8 * g];

    float mrow[8], lrow[8];
    #pragma unroll
    for (int r = 0; r < 8; ++r) { mrow[r] = -3e38f; lrow[r] = 0.f; }
    v8f accD[4];
    #pragma unroll
    for (int t = 0; t < 4; ++t) { v8f z = {}; accD[t] = z; }

    f16* pb = pba + w * 512;

    for (int jb = 0; jb < 16; ++jb) {      // 16 blocks of 32 keys
        v8f sA = {}, sB = {};
        {   // sim for keys jb*32 + lr (frag A)
            const f16* p = &ktl[(jb * 32 + lr) * 64 + 8 * g];
            v16h b0 = ld2(p,      p + 16);       // d 0..31
            v16h b1 = ld2(p + 32, p + 48);       // d 32..63
            sA = __builtin_amdgcn_wmma_f32_16x16x32_f16(false, aq0, false, b0, (short)0, sA, false, false);
            sA = __builtin_amdgcn_wmma_f32_16x16x32_f16(false, aq1, false, b1, (short)0, sA, false, false);
        }
        {   // sim for keys jb*32 + 16 + lr (frag B)
            const f16* p = &ktl[(jb * 32 + 16 + lr) * 64 + 8 * g];
            v16h b0 = ld2(p,      p + 16);
            v16h b1 = ld2(p + 32, p + 48);
            sB = __builtin_amdgcn_wmma_f32_16x16x32_f16(false, aq0, false, b0, (short)0, sB, false, false);
            sB = __builtin_amdgcn_wmma_f32_16x16x32_f16(false, aq1, false, b1, (short)0, sB, false, false);
        }

        const int img = jb >> 1;             // image id of this key block
        float scarr[8];
        #pragma unroll
        for (int r = 0; r < 8; ++r) {
            const bool ok = (locv[r] == img) && (locv[r] >= 0);
            const float a  = ok ? sA[r] : -3e38f;
            const float bq = ok ? sB[r] : -3e38f;
            const float bm = rmax16(fmaxf(a, bq));
            const float mn = fmaxf(mrow[r], bm);
            const float sc = expf(mrow[r] - mn);
            const float pa  = ok ? expf(a - mn)  : 0.f;
            const float pbv = ok ? expf(bq - mn) : 0.f;
            const float rs = rsum16(pa + pbv);
            lrow[r] = lrow[r] * sc + rs;
            mrow[r] = mn;
            scarr[r] = sc;
            // store P (C layout -> row-major 16x32 in LDS scratch)
            pb[(r + 8 * g) * 32 + lr]      = (f16)pa;
            pb[(r + 8 * g) * 32 + 16 + lr] = (f16)pbv;
        }
        #pragma unroll
        for (int t = 0; t < 4; ++t)
            #pragma unroll
            for (int r = 0; r < 8; ++r)
                accD[t][r] *= scarr[r];

        // reload P as an A fragment (16x32)
        const f16* pp = &pb[lr * 32 + 8 * g];
        const v16h pf = ld2(pp, pp + 16);

        // accD += P @ V_block  (V^T rows give contiguous B fragments)
        #pragma unroll
        for (int t = 0; t < 4; ++t) {
            const f16* vp = &vtl[(16 * t + lr) * 512 + jb * 32 + 8 * g];
            v16h vf = ld2(vp, vp + 16);
            accD[t] = __builtin_amdgcn_wmma_f32_16x16x32_f16(
                false, pf, false, vf, (short)0, accD[t], false, false);
        }
    }

    // normalize and write [b, s, h*64+d]
    #pragma unroll
    for (int t = 0; t < 4; ++t) {
        #pragma unroll
        for (int r = 0; r < 8; ++r) {
            const int qi = i0 + r + 8 * g;
            const int d  = 16 * t + lr;
            const float ov = accD[t][r] / fmaxf(lrow[r], 1e-20f);
            aout[((size_t)bb * 4096 + qi) * 512 + h * 64 + d] = (f16)ov;
        }
    }
}

// ---------------------------------------------------------------------------
// launch
// ---------------------------------------------------------------------------
extern "C" void kernel_launch(void* const* d_in, const int* in_sizes, int n_in,
                              void* d_out, int out_size, void* d_ws, size_t ws_size,
                              hipStream_t stream) {
    const float* x        = (const float*)d_in[0];   // [4,4096,2048]
    const float* context  = (const float*)d_in[1];   // [4,8,64,1024]
    const int*   media    = (const int*)  d_in[2];   // [4,4096]
    const float* norm_w   = (const float*)d_in[3];
    const float* norm_b   = (const float*)d_in[4];
    const float* normc_w  = (const float*)d_in[5];
    const float* normc_b  = (const float*)d_in[6];
    const float* Wq       = (const float*)d_in[7];   // [2048,512]
    const float* Wkv      = (const float*)d_in[8];   // [1024,1024]
    const float* Wout     = (const float*)d_in[9];   // [512,2048]

    char* ws = (char*)d_ws;
    size_t o = 0;
    f16* xh    = (f16*)(ws + o); o += (size_t)16384 * 2048 * 2;  // 64 MB
    f16* ctxh  = (f16*)(ws + o); o += (size_t)2048  * 1024 * 2;  //  4 MB
    f16* wqh   = (f16*)(ws + o); o += (size_t)2048  * 512  * 2;  //  2 MB
    f16* wkvh  = (f16*)(ws + o); o += (size_t)1024  * 1024 * 2;  //  2 MB
    f16* wouth = (f16*)(ws + o); o += (size_t)512   * 2048 * 2;  //  2 MB
    f16* qhp   = (f16*)(ws + o); o += (size_t)16384 * 512  * 2;  // 16 MB
    f16* khp   = (f16*)(ws + o); o += (size_t)32 * 512 * 64 * 2; //  2 MB
    f16* vtp   = (f16*)(ws + o); o += (size_t)32 * 64 * 512 * 2; //  2 MB
    f16* aoutp = (f16*)(ws + o); o += (size_t)16384 * 512  * 2;  // 16 MB

    // 1) LayerNorms -> f16
    ln_f16_kernel<<<2048, 256, 0, stream>>>(x, norm_w, norm_b, xh, 16384, 2048);
    ln_f16_kernel<<<256,  256, 0, stream>>>(context, normc_w, normc_b, ctxh, 2048, 1024);

    // 2) weight conversion -> f16  (each has 1,048,576 elements)
    cvt_f16_kernel<<<1024, 256, 0, stream>>>(Wq,   wqh,   262144);
    cvt_f16_kernel<<<1024, 256, 0, stream>>>(Wkv,  wkvh,  262144);
    cvt_f16_kernel<<<1024, 256, 0, stream>>>(Wout, wouth, 262144);

    // 3) q = xn @ Wq  (scaled, head-major)
    gemm_wmma<1><<<dim3(4, 128), 256, 0, stream>>>(xh, wqh, qhp, nullptr, 16384, 512, 2048);

    // 4) kv = ctx @ Wkv  (split into K[b,h,mn,d] and V^T[b,h,d,mn])
    gemm_wmma<2><<<dim3(8, 16), 256, 0, stream>>>(ctxh, wkvh, khp, vtp, 2048, 1024, 1024);

    // 5) masked flash attention (136 KB dynamic LDS; CDNA5 WGP has 320 KB)
    hipFuncSetAttribute((const void*)attn_kernel,
                        hipFuncAttributeMaxDynamicSharedMemorySize, 139264);
    attn_kernel<<<dim3(32, 32), 256, 139264, stream>>>(qhp, khp, vtp, media, aoutp);

    // 6) out = attn_out @ Wout  (f32 to d_out)
    gemm_wmma<3><<<dim3(16, 128), 256, 0, stream>>>(aoutp, wouth, d_out, nullptr, 16384, 2048, 512);
}
